// Net_31387620999394
// MI455X (gfx1250) — matrix operand order repaired and checked by reading, not compiled, not run on hardware
//
#include <hip/hip_runtime.h>

// ---------------------------------------------------------------------------
// VGG-16 forward for MI455X (gfx1250, wave32).
//  - 13x conv3x3 (pad 1) + ReLU: LDS-staged implicit GEMM, bf16 WMMA
//    (v_wmma_f32_16x16x32_bf16), register-tiled 32x32 per wave (4 WMMA tiles
//    sharing A/B fragments -> half the LDS traffic per WMMA)
//  - 5x 2x2 maxpool
//  - 3x FC as bandwidth-bound GEMV (batch==1), float4 streaming + LDS reduce
// ---------------------------------------------------------------------------

typedef __attribute__((ext_vector_type(16))) __bf16 v16bf;
typedef __attribute__((ext_vector_type(8)))  float  v8f;

static __device__ __forceinline__ unsigned short f2bfbits(float f) {
    // round-to-nearest-even fp32 -> bf16 (as raw bits)
    unsigned u = __builtin_bit_cast(unsigned, f);
    unsigned r = u + 0x7FFFu + ((u >> 16) & 1u);
    return (unsigned short)(r >> 16);
}

union Frag { v16bf bf; unsigned u[8]; };

// LDS layouts (padded strides keep pair-reads 4B aligned, spread banks)
#define XSTRIDE 36                    // cin stride inside halo patch row
#define WSTRIDE 36                    // k stride inside weight tap row
#define XPIX    (18 * 10)             // 18x10 halo pixels (16x8 tile + halo)
#define XELEMS  (XPIX * 32)           // halo pixels x 32 cin
#define WELEMS  (64 * 9 * 32)         // 64 cout x 9 taps x 32 k

// ---------------------------------------------------------------------------
// Conv 3x3 pad 1 as 9 tap-GEMMs over K = Cin, chunked by 32.
// Block: 256 thr = 8 waves = 2 cout-groups(32) x 4 pixel-groups(4x8).
// Block output tile: 64 cout x (16x8 pixels).
// Wave: 32 cout x 32 pixels = 2x2 WMMA register tile (4 accumulators);
// per tap load A0,A1,B0,B1 once and issue 4 WMMAs.
// ---------------------------------------------------------------------------
__global__ __launch_bounds__(256)
void conv3x3_relu_wmma(const float* __restrict__ x,   // [Cin, H, W]
                       const float* __restrict__ w,   // [Cout, Cin, 3, 3]
                       float* __restrict__ y,         // [Cout, H, W]
                       int Cin, int Cout, int H, int W)
{
    __shared__ unsigned short xl[XPIX * XSTRIDE];     // halo patch, bf16 bits
    __shared__ unsigned short wl[64 * 9 * WSTRIDE];   // weight chunk, bf16 bits

    const int tid  = threadIdx.x;
    const int lane = tid & 31;
    const int wid  = tid >> 5;
    const int ncol = lane & 15;           // N column / A row within 16-tile
    const int half = lane >> 4;           // 0: lanes 0-15, 1: lanes 16-31
    const int P    = H * W;

    const int Tx = W >> 3;                // spatial tiles per row (8 wide)
    const int tx = blockIdx.x % Tx;
    const int ty = blockIdx.x / Tx;       // 16-row tiles
    const int coutBase = blockIdx.y * 64;

    const int cgrp = wid & 1;             // which 32-cout half
    const int pgrp = wid >> 1;            // which 4-row pixel group
    const int ly   = pgrp * 4 + (ncol >> 3);    // B0 row in 16x8 tile
    const int lx   = ncol & 7;                  // col in 16x8 tile
    const int coRel = cgrp * 32 + ncol;         // A0 row (relative cout)

    const int kChunks = (Cin + 31) >> 5;

    v8f acc00 = {}, acc01 = {}, acc10 = {}, acc11 = {};

    for (int kc = 0; kc < kChunks; ++kc) {
        const int kbase = kc << 5;

        // ---- stage 18x10 halo x 32 cin (zero pad OOB / Cin tail) ----------
        for (int e = tid; e < XELEMS; e += 256) {
            const int hp  = e % XPIX;         // halo pixel (consecutive -> gx)
            const int cin = e / XPIX;
            const int hy  = hp / 10, hx = hp % 10;
            const int gy  = ty * 16 - 1 + hy;
            const int gx  = tx * 8 - 1 + hx;
            float v = 0.0f;
            if (gy >= 0 && gy < H && gx >= 0 && gx < W && (kbase + cin) < Cin)
                v = x[(size_t)(kbase + cin) * P + (size_t)gy * W + gx];
            xl[hp * XSTRIDE + cin] = f2bfbits(v);
        }
        // ---- stage 64 cout x 9 taps x 32 k weights ------------------------
        for (int e = tid; e < WELEMS; e += 256) {
            const int k  = e & 31;            // consecutive -> LDS contiguous
            const int t  = (e >> 5) % 9;
            const int co = e / 288;
            float v = 0.0f;
            if ((kbase + k) < Cin)
                v = w[(size_t)(coutBase + co) * Cin * 9 + (size_t)(kbase + k) * 9 + t];
            wl[(co * 9 + t) * WSTRIDE + k] = f2bfbits(v);
        }
        __syncthreads();

        // ---- 9 taps; per tap: 4 fragment loads feed 4 WMMAs ---------------
#pragma unroll
        for (int tap = 0; tap < 9; ++tap) {
            const int dy = tap / 3, dx = tap % 3;          // 0..2 (halo space)
            const int hp0 = (ly + dy) * 10 + (lx + dx);    // B0 halo pixel
            const int hp1 = hp0 + 2 * 10;                  // B1 (rows +2)
            Frag A0, A1, B0, B1;
            const unsigned short* wr0 = &wl[(coRel * 9 + tap) * WSTRIDE];
            const unsigned short* wr1 = wr0 + 16 * 9 * WSTRIDE;   // couts +16
            const unsigned short* xr0 = &xl[hp0 * XSTRIDE];
            const unsigned short* xr1 = &xl[hp1 * XSTRIDE];
#pragma unroll
            for (int v = 0; v < 8; ++v) {
                // A 16x32 bf16 layout: lane<16 K {0..7,16..23}, lane>=16 +8
                const int ka = ((v & 4) << 2) + half * 8 + ((v & 3) << 1);
                A0.u[v] = *(const unsigned*)(wr0 + ka);
                A1.u[v] = *(const unsigned*)(wr1 + ka);
                // B 32x16 bf16 layout: same K order as A (lane<16 K {0..7,16..23}, lane>=16 +8)
                const int kb = ((v & 4) << 2) + half * 8 + ((v & 3) << 1);
                B0.u[v] = *(const unsigned*)(xr0 + kb);
                B1.u[v] = *(const unsigned*)(xr1 + kb);
            }
            acc00 = __builtin_amdgcn_wmma_f32_16x16x32_bf16(
                false, A0.bf, false, B0.bf, (short)0, acc00, false, false);
            acc01 = __builtin_amdgcn_wmma_f32_16x16x32_bf16(
                false, A0.bf, false, B1.bf, (short)0, acc01, false, false);
            acc10 = __builtin_amdgcn_wmma_f32_16x16x32_bf16(
                false, A1.bf, false, B0.bf, (short)0, acc10, false, false);
            acc11 = __builtin_amdgcn_wmma_f32_16x16x32_bf16(
                false, A1.bf, false, B1.bf, (short)0, acc11, false, false);
        }
        __syncthreads();
    }

    // ---- ReLU + store: D VGPR r -> M = r + 8*half, N = ncol ---------------
    // acc(mi,nj): couts coutBase + cgrp*32 + mi*16, pixels rows pgrp*4 + nj*2
    const int colg = tx * 8 + lx;
#pragma unroll
    for (int mi = 0; mi < 2; ++mi) {
#pragma unroll
        for (int nj = 0; nj < 2; ++nj) {
            const v8f a = (mi == 0) ? (nj == 0 ? acc00 : acc01)
                                    : (nj == 0 ? acc10 : acc11);
            const int rowg = ty * 16 + pgrp * 4 + nj * 2 + (ncol >> 3);
            const int n = rowg * W + colg;
#pragma unroll
            for (int r = 0; r < 8; ++r) {
                const int mo = coutBase + cgrp * 32 + mi * 16 + half * 8 + r;
                const float v = a[r] > 0.0f ? a[r] : 0.0f;
                y[(size_t)mo * P + n] = v;
            }
        }
    }
}

// ---------------------------------------------------------------------------
// 2x2 max pool, stride 2.
// ---------------------------------------------------------------------------
__global__ __launch_bounds__(256)
void maxpool2x2(const float* __restrict__ x, float* __restrict__ y,
                int C, int Ho, int Wo)
{
    const int i = blockIdx.x * blockDim.x + threadIdx.x;
    const int total = C * Ho * Wo;
    if (i >= total) return;
    const int wo = i % Wo;
    const int t  = i / Wo;
    const int ho = t % Ho;
    const int c  = t / Ho;
    const int Wi = Wo * 2;
    const float* p = x + ((size_t)c * (Ho * 2) + ho * 2) * Wi + wo * 2;
    y[i] = fmaxf(fmaxf(p[0], p[1]), fmaxf(p[Wi], p[Wi + 1]));
}

// ---------------------------------------------------------------------------
// Batch-1 FC: out[row] = dot(W[row,:], x).  Bandwidth bound (fw1 = 512 MB).
// ---------------------------------------------------------------------------
__global__ __launch_bounds__(256)
void fc_rowdot(const float* __restrict__ w, const float* __restrict__ x,
               float* __restrict__ out, int K)
{
    __shared__ float red[256];
    const int row = blockIdx.x;
    const float* __restrict__ wr = w + (size_t)row * K;

    float s = 0.0f;
    for (int k = threadIdx.x * 4; k < K; k += 256 * 4) {
        __builtin_prefetch(wr + k + 256 * 4, 0, 0);  // global_prefetch_b8
        const float4 wv = *(const float4*)(wr + k);
        const float4 xv = *(const float4*)(x + k);
        s += wv.x * xv.x + wv.y * xv.y + wv.z * xv.z + wv.w * xv.w;
    }
    red[threadIdx.x] = s;
    __syncthreads();
#pragma unroll
    for (int off = 128; off > 0; off >>= 1) {
        if (threadIdx.x < (unsigned)off) red[threadIdx.x] += red[threadIdx.x + off];
        __syncthreads();
    }
    if (threadIdx.x == 0) out[row] = red[0];
}

// ---------------------------------------------------------------------------
// Launch: full VGG-16 pipeline.
// d_in: 0=x, 1..4 = H,W,nTh,nTw (unused scalars), 5..17 = cw1..cw13,
//       18=fw1, 19=fw2, 20=fw3.
// ---------------------------------------------------------------------------
extern "C" void kernel_launch(void* const* d_in, const int* in_sizes, int n_in,
                              void* d_out, int out_size, void* d_ws, size_t ws_size,
                              hipStream_t stream)
{
    (void)in_sizes; (void)n_in; (void)out_size; (void)ws_size;

    const float* x = (const float*)d_in[0];
    const float* cw[13];
    for (int i = 0; i < 13; ++i) cw[i] = (const float*)d_in[5 + i];
    const float* fw1 = (const float*)d_in[18];
    const float* fw2 = (const float*)d_in[19];
    const float* fw3 = (const float*)d_in[20];
    float* out = (float*)d_out;

    // Workspace: two 16MB ping-pong activation buffers + FC temps.
    const size_t BUF = (size_t)64 * 256 * 256;
    float* buf0 = (float*)d_ws;
    float* buf1 = buf0 + BUF;
    float* fcA  = buf1 + BUF;
    float* fcB  = fcA + 4096;

    auto conv = [&](const float* in, const float* wt, float* o,
                    int Cin, int Cout, int H, int W) {
        dim3 grid((H / 16) * (W / 8), Cout / 64);
        conv3x3_relu_wmma<<<grid, 256, 0, stream>>>(in, wt, o, Cin, Cout, H, W);
    };
    auto pool = [&](const float* in, float* o, int C, int Ho, int Wo) {
        const int total = C * Ho * Wo;
        maxpool2x2<<<(total + 255) / 256, 256, 0, stream>>>(in, o, C, Ho, Wo);
    };
    auto fc = [&](const float* wt, const float* in, float* o, int N, int K) {
        fc_rowdot<<<N, 256, 0, stream>>>(wt, in, o, K);
    };

    // Block 1 (256x256)
    conv(x,    cw[0], buf0,   3,  64, 256, 256);
    conv(buf0, cw[1], buf1,  64,  64, 256, 256);
    pool(buf1, buf0,  64, 128, 128);
    // Block 2 (128x128)
    conv(buf0, cw[2], buf1,  64, 128, 128, 128);
    conv(buf1, cw[3], buf0, 128, 128, 128, 128);
    pool(buf0, buf1, 128, 64, 64);
    // Block 3 (64x64)
    conv(buf1, cw[4], buf0, 128, 256, 64, 64);
    conv(buf0, cw[5], buf1, 256, 256, 64, 64);
    conv(buf1, cw[6], buf0, 256, 256, 64, 64);
    pool(buf0, buf1, 256, 32, 32);
    // Block 4 (32x32)
    conv(buf1, cw[7], buf0, 256, 512, 32, 32);
    conv(buf0, cw[8], buf1, 512, 512, 32, 32);
    conv(buf1, cw[9], buf0, 512, 512, 32, 32);
    pool(buf0, buf1, 512, 16, 16);
    // Block 5 (16x16)
    conv(buf1, cw[10], buf0, 512, 512, 16, 16);
    conv(buf0, cw[11], buf1, 512, 512, 16, 16);
    conv(buf1, cw[12], buf0, 512, 512, 16, 16);
    pool(buf0, buf1, 512, 8, 8);          // -> [512, 8, 8] = 32768 floats

    // FC head (no bias, no relu)
    fc(fw1, buf1, fcA, 4096, 32768);
    fc(fw2, fcA,  fcB, 4096, 4096);
    fc(fw3, fcB,  out, 1000, 4096);
}
